// MultiHeadAttention_8564164788990
// MI455X (gfx1250) — matrix operand order repaired and checked by reading, not compiled, not run on hardware
//
#include <hip/hip_runtime.h>

#define Bsz 8
#define Lseq 512
#define Dm 512
#define Hn 8
#define Dh 64

typedef _Float16 half16 __attribute__((ext_vector_type(16)));
typedef float    float8 __attribute__((ext_vector_type(8)));

union H16 { half16 h; uint4 u[2]; };

__device__ __forceinline__ float8 wmma32(half16 a, half16 b, float8 c) {
  // v_wmma_f32_16x16x32_f16 : D = A(16x32 f16) * B(32x16 f16) + C(16x16 f32)
  return __builtin_amdgcn_wmma_f32_16x16x32_f16(false, a, false, b, (short)0, c,
                                                false, false);
}

// A-fragment (16x32, f16, row-major source, row stride ld elements).
// ISA layout: lanes 0-15 = rows M, elems 0..7 -> K 0..7, elems 8..15 -> K 16..23
//             lanes 16-31 = same rows, K 8..15 and K 24..31.
__device__ __forceinline__ half16 load_frag_a(const _Float16* p, int ld, int lane) {
  const int m  = lane & 15;
  const int hi = lane >> 4;
  const _Float16* r0 = p + (size_t)m * ld + hi * 8;
  H16 t;
  t.u[0] = *(const uint4*)(r0);        // K base+0..7   (or +8..15)
  t.u[1] = *(const uint4*)(r0 + 16);   // K base+16..23 (or +24..31)
  return t.h;
}

// B-fragment (32x16, f16). Column n of B must be contiguous in memory at
// p + n*ld (i.e. source is the row-major matrix whose row n is column n of B).
// ISA layout: lanes 0-15 = col n, K 0..7 and 16..23; lanes 16-31 = col n, K 8..15 and 24..31.
__device__ __forceinline__ half16 load_frag_b(const _Float16* p, int ld, int lane) {
  const int n  = lane & 15;
  const int hi = lane >> 4;
  const _Float16* r0 = p + (size_t)n * ld + hi * 8;
  H16 t;
  t.u[0] = *(const uint4*)(r0);
  t.u[1] = *(const uint4*)(r0 + 16);
  return t.h;
}

__global__ void cvt_f32_to_f16_kernel(const float* __restrict__ src,
                                      _Float16* __restrict__ dst, int n) {
  int i = blockIdx.x * blockDim.x + threadIdx.x;
  if (i < n) dst[i] = (_Float16)src[i];
}

// bias[q][k] = sum_d rel_bias[q][k][d] ; one wave per (q,k) row of 512 floats.
__global__ void bias_reduce_kernel(const float* __restrict__ rel,
                                   float* __restrict__ biasLL) {
  const int gw   = (blockIdx.x * blockDim.x + threadIdx.x) >> 5;
  const int lane = threadIdx.x & 31;
  const float4* row = (const float4*)(rel + (size_t)gw * 512);
  float s = 0.f;
#pragma unroll
  for (int i = 0; i < 4; ++i) {
    float4 v = row[lane + 32 * i];
    s += v.x + v.y + v.z + v.w;
  }
#pragma unroll
  for (int o = 16; o; o >>= 1) s += __shfl_xor(s, o, 32);
  if (lane == 0) biasLL[gw] = s;
}

// One wave: 16 tokens x one head (64 dims) of Q, K or V. GEMM via WMMA,
// then RoPE (Q,K) exactly as the reference: out[j]    = X[2j]c - X[2j+1]s,
//                                           out[j+32] = X[2j]s + X[2j+1]c.
__global__ void qkv_rope_kernel(const _Float16* __restrict__ xh,
                                const _Float16* __restrict__ Wqh,
                                const _Float16* __restrict__ Wkh,
                                const _Float16* __restrict__ Wvh,
                                const float* __restrict__ bq,
                                const float* __restrict__ bk,
                                const float* __restrict__ bv,
                                _Float16* __restrict__ Qh,   // [b][h][l][dh]
                                _Float16* __restrict__ Kh,   // [b][h][l][dh]
                                _Float16* __restrict__ Vt) { // [b][h][dh][l]
  const int lane  = threadIdx.x;
  const int nbase = blockIdx.x * 16;   // token tile (tokens = b*L + l)
  const int h     = blockIdx.y;
  const int pz    = blockIdx.z;        // 0=Q 1=K 2=V

  const _Float16* W  = (pz == 0) ? Wqh : (pz == 1) ? Wkh : Wvh;
  const float* bvec  = (pz == 0) ? bq  : (pz == 1) ? bk  : bv;

  float8 acc[4] = {};
  for (int kb = 0; kb < Dm; kb += 32) {
    half16 a = load_frag_a(xh + (size_t)nbase * Dm + kb, Dm, lane);
#pragma unroll
    for (int t = 0; t < 4; ++t) {
      half16 bf = load_frag_b(W + (size_t)(h * Dh + t * 16) * Dm + kb, Dm, lane);
      acc[t] = wmma32(a, bf, acc[t]);
    }
  }

  __shared__ float raw[16][Dh];
  const int col = lane & 15;
  const int hi  = lane >> 4;
#pragma unroll
  for (int t = 0; t < 4; ++t)
#pragma unroll
    for (int v = 0; v < 8; ++v)
      raw[v + 8 * hi][t * 16 + col] = acc[t][v] + bvec[h * Dh + t * 16 + col];
  __syncthreads();

  for (int idx = lane; idx < 16 * Dh; idx += 32) {
    const int r   = idx >> 6;
    const int j   = idx & 63;
    const int n   = nbase + r;
    const int bb  = n >> 9;            // L = 512
    const int pos = n & (Lseq - 1);
    if (pz == 2) {
      Vt[((size_t)(bb * Hn + h) * Dh + j) * Lseq + pos] = (_Float16)raw[r][j];
    } else {
      const int jp  = (j < 32) ? j : j - 32;
      const float freq = __expf(-0.28782313662425572f * (float)jp); // 10000^(-jp/32)
      const float ang  = (float)pos * freq;
      float sn, cs;
      __sincosf(ang, &sn, &cs);
      const float x1 = raw[r][2 * jp];
      const float x2 = raw[r][2 * jp + 1];
      const float outv = (j < 32) ? (x1 * cs - x2 * sn) : (x1 * sn + x2 * cs);
      _Float16* dst = (pz == 0) ? Qh : Kh;
      dst[((size_t)(bb * Hn + h) * Lseq + pos) * Dh + j] = (_Float16)outv;
    }
  }
}

// One wave: one (b,h) and one 16-row q tile. Full 512-key score row in LDS.
__global__ void attn_kernel(const _Float16* __restrict__ Qh,
                            const _Float16* __restrict__ Kh,
                            const _Float16* __restrict__ Vt,
                            const float* __restrict__ biasLL,
                            _Float16* __restrict__ ctxh) { // [b][h][l][dh]
  const int lane  = threadIdx.x;
  const int qt    = blockIdx.x;
  const int h     = blockIdx.y;
  const int b     = blockIdx.z;
  const size_t bh = (size_t)b * Hn + h;
  const int qbase = qt * 16;

  __shared__ float sc[16][Lseq];   // 32 KB
  __shared__ float rowinv[16];

  const _Float16* Qp = Qh + (bh * Lseq + qbase) * Dh;
  half16 a0 = load_frag_a(Qp, Dh, lane);       // dh 0..31
  half16 a1 = load_frag_a(Qp + 32, Dh, lane);  // dh 32..63

  const int col = lane & 15;
  const int hi  = lane >> 4;

  for (int kt = 0; kt < Lseq; kt += 16) {
    float8 acc = {};
    const _Float16* Kp = Kh + (bh * Lseq + kt) * Dh;
    half16 b0 = load_frag_b(Kp, Dh, lane);
    half16 b1 = load_frag_b(Kp + 32, Dh, lane);
    acc = wmma32(a0, b0, acc);
    acc = wmma32(a1, b1, acc);
#pragma unroll
    for (int v = 0; v < 8; ++v) {
      const int M = v + 8 * hi;
      sc[M][kt + col] =
          acc[v] * 0.125f + biasLL[(size_t)(qbase + M) * Lseq + kt + col];
    }
  }
  __syncthreads();

  for (int r = 0; r < 16; ++r) {
    float vals[16];
    float mx = -3.0e38f;
#pragma unroll
    for (int i = 0; i < 16; ++i) {
      vals[i] = sc[r][lane + 32 * i];
      mx = fmaxf(mx, vals[i]);
    }
#pragma unroll
    for (int o = 16; o; o >>= 1) mx = fmaxf(mx, __shfl_xor(mx, o, 32));
    float sum = 0.f;
#pragma unroll
    for (int i = 0; i < 16; ++i) {
      const float e = __expf(vals[i] - mx);
      sum += e;
      sc[r][lane + 32 * i] = e;   // write exp back in place
    }
#pragma unroll
    for (int o = 16; o; o >>= 1) sum += __shfl_xor(sum, o, 32);
    if (lane == 0) rowinv[r] = 1.0f / sum;
  }
  __syncthreads();

  float8 acc[4] = {};
  const float rinv = rowinv[col];      // this lane's A-matrix row
  for (int kc = 0; kc < Lseq; kc += 32) {
    half16 a;
#pragma unroll
    for (int e = 0; e < 16; ++e) {
      const int koff = ((e < 8) ? e : (e + 8)) + hi * 8;   // ISA A layout
      a[e] = (_Float16)(sc[col][kc + koff] * rinv);
    }
#pragma unroll
    for (int t = 0; t < 4; ++t) {
      half16 bf = load_frag_b(Vt + (bh * Dh + t * 16) * Lseq + kc, Lseq, lane);
      acc[t] = wmma32(a, bf, acc[t]);
    }
  }
#pragma unroll
  for (int t = 0; t < 4; ++t)
#pragma unroll
    for (int v = 0; v < 8; ++v)
      ctxh[(bh * Lseq + qbase + v + 8 * hi) * Dh + t * 16 + col] =
          (_Float16)(acc[t][v]);
}

// out = ctx_flat @ Wo.T + bo  (ctx_flat[n][h*64+d] = ctxh[b][h][l][d])
__global__ void out_proj_kernel(const _Float16* __restrict__ ctxh,
                                const _Float16* __restrict__ Woh,
                                const float* __restrict__ bo,
                                float* __restrict__ out) {
  const int lane  = threadIdx.x;
  const int nbase = blockIdx.x * 16;
  const int mbase = blockIdx.y * 64;
  const int bb    = nbase >> 9;
  const int lbase = nbase & (Lseq - 1);
  const int col   = lane & 15;
  const int hi    = lane >> 4;

  float8 acc[4] = {};
  for (int kb = 0; kb < Dm; kb += 32) {
    const int hh = kb >> 6;       // head (K-chunks never cross a head: 64%32==0)
    const int d0 = kb & 63;
    half16 a = load_frag_a(
        ctxh + ((size_t)(bb * Hn + hh) * Lseq + lbase) * Dh + d0, Dh, lane);
#pragma unroll
    for (int t = 0; t < 4; ++t) {
      half16 bf = load_frag_b(Woh + (size_t)(mbase + t * 16) * Dm + kb, Dm, lane);
      acc[t] = wmma32(a, bf, acc[t]);
    }
  }
#pragma unroll
  for (int t = 0; t < 4; ++t)
#pragma unroll
    for (int v = 0; v < 8; ++v) {
      const int M = v + 8 * hi;
      const int m = mbase + t * 16 + col;
      out[(size_t)(nbase + M) * Dm + m] = acc[t][v] + bo[m];
    }
}

extern "C" void kernel_launch(void* const* d_in, const int* in_sizes, int n_in,
                              void* d_out, int out_size, void* d_ws, size_t ws_size,
                              hipStream_t stream) {
  (void)in_sizes; (void)n_in; (void)out_size; (void)ws_size;
  const float* x   = (const float*)d_in[0];
  const float* rel = (const float*)d_in[1];
  const float* Wq  = (const float*)d_in[2];
  const float* bq  = (const float*)d_in[3];
  const float* Wk  = (const float*)d_in[4];
  const float* bk  = (const float*)d_in[5];
  const float* Wv  = (const float*)d_in[6];
  const float* bv  = (const float*)d_in[7];
  const float* Wo  = (const float*)d_in[8];
  const float* bo  = (const float*)d_in[9];
  float* out = (float*)d_out;

  char* ws = (char*)d_ws;
  size_t off = 0;
  float*    biasLL = (float*)(ws + off);    off += (size_t)Lseq * Lseq * 4;      // 1 MB
  _Float16* xh     = (_Float16*)(ws + off); off += (size_t)Bsz * Lseq * Dm * 2;  // 4 MB
  _Float16* Wqh    = (_Float16*)(ws + off); off += (size_t)Dm * Dm * 2;
  _Float16* Wkh    = (_Float16*)(ws + off); off += (size_t)Dm * Dm * 2;
  _Float16* Wvh    = (_Float16*)(ws + off); off += (size_t)Dm * Dm * 2;
  _Float16* Woh    = (_Float16*)(ws + off); off += (size_t)Dm * Dm * 2;
  _Float16* Qh     = (_Float16*)(ws + off); off += (size_t)Bsz * Hn * Lseq * Dh * 2;
  _Float16* Kh     = (_Float16*)(ws + off); off += (size_t)Bsz * Hn * Lseq * Dh * 2;
  _Float16* Vt     = (_Float16*)(ws + off); off += (size_t)Bsz * Hn * Lseq * Dh * 2;
  _Float16* ctxh   = (_Float16*)(ws + off); off += (size_t)Bsz * Hn * Lseq * Dh * 2;

  const int nx = Bsz * Lseq * Dm;   // 2,097,152
  const int nw = Dm * Dm;           //   262,144
  cvt_f32_to_f16_kernel<<<(nx + 255) / 256, 256, 0, stream>>>(x, xh, nx);
  cvt_f32_to_f16_kernel<<<(nw + 255) / 256, 256, 0, stream>>>(Wq, Wqh, nw);
  cvt_f32_to_f16_kernel<<<(nw + 255) / 256, 256, 0, stream>>>(Wk, Wkh, nw);
  cvt_f32_to_f16_kernel<<<(nw + 255) / 256, 256, 0, stream>>>(Wv, Wvh, nw);
  cvt_f32_to_f16_kernel<<<(nw + 255) / 256, 256, 0, stream>>>(Wo, Woh, nw);

  // 512*512 rows, one wave each, 8 waves per 256-thread block
  bias_reduce_kernel<<<(Lseq * Lseq) / 8, 256, 0, stream>>>(rel, biasLL);

  qkv_rope_kernel<<<dim3(Bsz * Lseq / 16, Hn, 3), 32, 0, stream>>>(
      xh, Wqh, Wkh, Wvh, bq, bk, bv, Qh, Kh, Vt);

  attn_kernel<<<dim3(Lseq / 16, Hn, Bsz), 32, 0, stream>>>(
      Qh, Kh, Vt, biasLL, ctxh);

  out_proj_kernel<<<dim3(Bsz * Lseq / 16, Dm / 64), 32, 0, stream>>>(
      ctxh, Woh, bo, out);
}
